// ChebyNet_10548439679261
// MI455X (gfx1250) — matrix operand order repaired and checked by reading, not compiled, not run on hardware
//
#include <hip/hip_runtime.h>
#include <hip/hip_bf16.h>

// ---------------------------------------------------------------------------
// ChebyNet forward for MI455X (gfx1250).
// Sparse scatter: one wave per edge, f32 atomics (HBM-bound, ~160us floor).
// Dense GEMMs: v_wmma_f32_16x16x32_bf16, register-blocked 4 column tiles per
// wave (32 accumulator VGPRs -> no spills), K=128 compile-time, fully unrolled.
// ---------------------------------------------------------------------------

#define NROI   116
#define NNODES 59392          // 512 * 116
#define NEDGES 1900544        // NNODES * 32
#define BATCH  512
#define K1PAD  128            // 116 padded to 128 for WMMA K (both layers K=128)
#define EMB0   128
#define EMB1   64

typedef __attribute__((ext_vector_type(16))) __bf16 v16bf;
typedef __attribute__((ext_vector_type(8)))  __bf16 v8bf;
typedef __attribute__((ext_vector_type(8)))  float  v8f;

__device__ __forceinline__ float mishf(float x) {
    float sp = (x > 20.f) ? x : log1pf(expf(x));   // softplus, stable
    return x * tanhf(sp);
}

// ---------------------------------------------------------------- utilities
__global__ void k_zero(float* p, int n) {
    int i = blockIdx.x * blockDim.x + threadIdx.x;
    if (i < n) p[i] = 0.f;
}

__global__ void k_deg(const int* __restrict__ src, const float* __restrict__ w,
                      float* __restrict__ deg, int e) {
    int i = blockIdx.x * blockDim.x + threadIdx.x;
    if (i < e) atomicAdd(&deg[src[i]], w[i]);
}

__global__ void k_dis(const float* __restrict__ deg, float* __restrict__ dis, int n) {
    int i = blockIdx.x * blockDim.x + threadIdx.x;
    if (i < n) {
        float d = deg[i];
        dis[i] = (d > 0.f) ? rsqrtf(fmaxf(d, 1e-30f)) : 0.f;
    }
}

__global__ void k_norm(const int* __restrict__ src, const int* __restrict__ dst,
                       const float* __restrict__ w, const float* __restrict__ dis,
                       float* __restrict__ norm, int e) {
    int i = blockIdx.x * blockDim.x + threadIdx.x;
    if (i < e) norm[i] = -dis[dst[i]] * w[i] * dis[src[i]];
}

// x [N,116] f32 -> xpad [N,128] bf16 (zero padded)
__global__ void k_xpad(const float* __restrict__ x, __bf16* __restrict__ xp, int n) {
    int i = blockIdx.x * blockDim.x + threadIdx.x;
    if (i >= n * K1PAD) return;
    int r = i >> 7, c = i & 127;
    xp[i] = (__bf16)((c < NROI) ? x[r * NROI + c] : 0.f);
}

// f32 [n] -> bf16 [n]
__global__ void k_f2bf(const float* __restrict__ in, __bf16* __restrict__ out, int n) {
    int i = blockIdx.x * blockDim.x + threadIdx.x;
    if (i < n) out[i] = (__bf16)in[i];
}

// Pack weight W [Ksrc, Nout] f32 row-major into WMMA B-fragment order (bf16).
// Fragment (kt,nt): 32 lanes x 16 values; lane<16 -> col nt*16+lane, K kt*32+{0..7,16..23}
//                                          lane>=16 -> col nt*16+lane-16, K kt*32+{8..15,24..31}
__global__ void k_packw(const float* __restrict__ W, __bf16* __restrict__ out,
                        int ksrc, int kpad, int nout) {
    int nt_cnt = nout >> 4;
    int tid = blockIdx.x * blockDim.x + threadIdx.x;          // one thread = one lane
    int total = (kpad >> 5) * nt_cnt * 32;
    if (tid >= total) return;
    int lane = tid & 31;
    int frag = tid >> 5;
    int nt = frag % nt_cnt, kt = frag / nt_cnt;
    int col = nt * 16 + (lane & 15);
    int kb  = kt * 32 + ((lane >> 4) << 3);
    #pragma unroll
    for (int i = 0; i < 16; ++i) {
        int k = kb + (i < 8 ? i : i + 8);
        out[(size_t)tid * 16 + i] = (__bf16)((k < ksrc) ? W[(size_t)k * nout + col] : 0.f);
    }
}

// ------------------------------------------------------------- edge scatter
// One wave per edge: lane L handles features 4L..4L+3. src/dst/norm are
// wave-broadcast loads; the feature row read is fully coalesced.
// layer 1: acc[dst,0:116] += norm * x[src,0:116]   (acc stride 128, f32 src)
__global__ void k_scatter_l1(const int* __restrict__ src, const int* __restrict__ dst,
                             const float* __restrict__ norm, const float* __restrict__ x,
                             float* __restrict__ acc) {
    int e    = (blockIdx.x * blockDim.x + threadIdx.x) >> 5;  // wave id
    int lane = threadIdx.x & 31;
    if (e >= NEDGES) return;
    int s = src[e], d = dst[e];
    float w = norm[e];
    int f = lane * 4;
    if (f >= NROI) return;                                    // lanes 29..31 idle
    const float* xs = x + (size_t)s * NROI + f;
    float* ad = acc + (size_t)d * K1PAD + f;
    #pragma unroll
    for (int j = 0; j < 4; ++j) atomicAdd(&ad[j], w * xs[j]);
}

// layer 2: acc[dst,0:128] += norm * h1[src,0:128]  (bf16 source)
__global__ void k_scatter_l2(const int* __restrict__ src, const int* __restrict__ dst,
                             const float* __restrict__ norm, const __bf16* __restrict__ h,
                             float* __restrict__ acc) {
    int e    = (blockIdx.x * blockDim.x + threadIdx.x) >> 5;  // wave id
    int lane = threadIdx.x & 31;
    if (e >= NEDGES) return;
    int s = src[e], d = dst[e];
    float w = norm[e];
    int f = lane * 4;
    const __bf16* hs = h + (size_t)s * EMB0 + f;
    float* ad = acc + (size_t)d * EMB0 + f;
    #pragma unroll
    for (int j = 0; j < 4; ++j) atomicAdd(&ad[j], w * (float)hs[j]);
}

// ---------------------------------------------------------- fused dual GEMM
// D[M, NTOT*16] = mish( A0[M,128]@B0 + A1[M,128]@B1 + bias ).
// One wave per (16-row M block, group of NT column tiles). NT=4 keeps the
// accumulator file at 32 VGPRs so the fully unrolled K loop never spills;
// each A fragment feeds NT WMMAs.
template <int NT, int NTOT>
__global__ void k_gemm_dual(const __bf16* __restrict__ A0, const __bf16* __restrict__ A1,
                            const __bf16* __restrict__ B0, const __bf16* __restrict__ B1,
                            const float* __restrict__ bias, int M,
                            __bf16* __restrict__ outb, float* __restrict__ outf) {
    constexpr int K      = 128;
    constexpr int KT     = K / 32;
    constexpr int Nout   = NTOT * 16;
    constexpr int GROUPS = NTOT / NT;

    int wv   = (blockIdx.x * blockDim.x + threadIdx.x) >> 5;  // global wave id
    int lane = threadIdx.x & 31;
    if (wv >= (M >> 4) * GROUPS) return;                      // wave-uniform exit
    int mt  = wv / GROUPS;
    int nt0 = (wv % GROUPS) * NT;

    int rowA = mt * 16 + (lane & 15);
    int koff = (lane >> 4) << 3;                              // A layout: 0 or 8
    const __bf16* a0r = A0 + (size_t)rowA * K;
    const __bf16* a1r = A1 + (size_t)rowA * K;

    v8f c[NT];
    #pragma unroll
    for (int nt = 0; nt < NT; ++nt)
        c[nt] = (v8f){0.f, 0.f, 0.f, 0.f, 0.f, 0.f, 0.f, 0.f};

    #pragma unroll
    for (int kt = 0; kt < KT; ++kt) {
        int kb = kt * 32 + koff;
        union { v16bf v; v8bf h[2]; } ua;
        // ---- A0 fragment: K {kb..kb+7, kb+16..kb+23} ----
        ua.h[0] = *(const v8bf*)(a0r + kb);
        ua.h[1] = *(const v8bf*)(a0r + kb + 16);
        #pragma unroll
        for (int nt = 0; nt < NT; ++nt) {
            v16bf bf = *(const v16bf*)(B0 + (((size_t)kt * NTOT + nt0 + nt) * 32 + lane) * 16);
            c[nt] = __builtin_amdgcn_wmma_f32_16x16x32_bf16(false, ua.v, false, bf,
                                                            (short)0, c[nt], false, false);
        }
        // ---- A1 fragment ----
        ua.h[0] = *(const v8bf*)(a1r + kb);
        ua.h[1] = *(const v8bf*)(a1r + kb + 16);
        #pragma unroll
        for (int nt = 0; nt < NT; ++nt) {
            v16bf bf = *(const v16bf*)(B1 + (((size_t)kt * NTOT + nt0 + nt) * 32 + lane) * 16);
            c[nt] = __builtin_amdgcn_wmma_f32_16x16x32_bf16(false, ua.v, false, bf,
                                                            (short)0, c[nt], false, false);
        }
    }

    int rb = mt * 16 + ((lane >> 4) << 3);                    // D layout: lanes>=16 -> M+8
    #pragma unroll
    for (int nt = 0; nt < NT; ++nt) {
        int col = (nt0 + nt) * 16 + (lane & 15);
        float bv = bias[col];
        #pragma unroll
        for (int i = 0; i < 8; ++i) {
            float v = mishf(c[nt][i] + bv);
            size_t idx = (size_t)(rb + i) * Nout + col;
            if (outb) outb[idx] = (__bf16)v;
            if (outf) outf[idx] = v;
        }
    }
}

// ----------------------------------------------------------------- MLP tail
__global__ void k_readout(const float* __restrict__ h2, const float* __restrict__ w,
                          const float* __restrict__ b, float* __restrict__ out) {
    int tid = blockIdx.x * blockDim.x + threadIdx.x;
    if (tid >= NNODES * 8) return;
    int j = tid & 7, nn = tid >> 3;
    const float* hr = h2 + (size_t)nn * EMB1;
    float s = b[j];
    #pragma unroll 8
    for (int k = 0; k < EMB1; ++k) s += hr[k] * w[k * 8 + j];
    out[tid] = mishf(s);
}

__global__ void k_fc1(const float* __restrict__ feats, const float* __restrict__ w,
                      const float* __restrict__ b, float* __restrict__ z) {
    int tid = blockIdx.x * blockDim.x + threadIdx.x;
    if (tid >= BATCH * NROI) return;
    int c = tid % NROI, bb = tid / NROI;
    const float* fr = feats + (size_t)bb * (NROI * 8);
    float s = b[c];
    for (int i = 0; i < NROI * 8; ++i) s += fr[i] * w[(size_t)i * NROI + c];
    z[tid] = s;
}

__global__ void k_bnstats(const float* __restrict__ z, float* __restrict__ mu,
                          float* __restrict__ var) {
    int c = blockIdx.x, t = threadIdx.x;              // 116 blocks x 256 threads
    float s = 0.f, q = 0.f;
    for (int b = t; b < BATCH; b += 256) {
        float v = z[b * NROI + c];
        s += v; q += v * v;
    }
    __shared__ float ss[256], sq[256];
    ss[t] = s; sq[t] = q; __syncthreads();
    for (int o = 128; o > 0; o >>= 1) {
        if (t < o) { ss[t] += ss[t + o]; sq[t] += sq[t + o]; }
        __syncthreads();
    }
    if (t == 0) {
        float m = ss[0] / (float)BATCH;
        mu[c] = m;
        var[c] = sq[0] / (float)BATCH - m * m;
    }
}

__global__ void k_final(const float* __restrict__ z, const float* __restrict__ mu,
                        const float* __restrict__ var, const float* __restrict__ g,
                        const float* __restrict__ bb, const float* __restrict__ w2,
                        const float* __restrict__ b2, float* __restrict__ out) {
    int b = blockIdx.x * blockDim.x + threadIdx.x;
    if (b >= BATCH) return;
    float a0 = b2[0], a1 = b2[1];
    for (int c = 0; c < NROI; ++c) {
        float zn = (z[b * NROI + c] - mu[c]) * rsqrtf(var[c] + 1e-5f) * g[c] + bb[c];
        float m = mishf(zn);
        a0 += m * w2[c * 2 + 0];
        a1 += m * w2[c * 2 + 1];
    }
    out[b * 2 + 0] = a0;
    out[b * 2 + 1] = a1;
}

// ---------------------------------------------------------------------------
extern "C" void kernel_launch(void* const* d_in, const int* in_sizes, int n_in,
                              void* d_out, int out_size, void* d_ws, size_t ws_size,
                              hipStream_t stream) {
    const float* x     = (const float*)d_in[0];
    const int*   ei    = (const int*)d_in[1];          // [2,E] int32
    const float* eattr = (const float*)d_in[2];
    const float* w1_0  = (const float*)d_in[3];
    const float* w1_1  = (const float*)d_in[4];
    const float* b1    = (const float*)d_in[5];
    const float* w2_0  = (const float*)d_in[6];
    const float* w2_1  = (const float*)d_in[7];
    const float* b2    = (const float*)d_in[8];
    const float* w_ro  = (const float*)d_in[9];
    const float* b_ro  = (const float*)d_in[10];
    const float* w_fc1 = (const float*)d_in[11];
    const float* b_fc1 = (const float*)d_in[12];
    const float* bn_g  = (const float*)d_in[13];
    const float* bn_b  = (const float*)d_in[14];
    const float* w_fc2 = (const float*)d_in[15];
    const float* b_fc2 = (const float*)d_in[16];
    const int* esrc = ei;
    const int* edst = ei + NEDGES;

    // ---- workspace carve (256B aligned) ------------------------------------
    char* ws = (char*)d_ws;
    size_t off = 0;
    auto carve = [&](size_t bytes) {
        char* p = ws + off;
        off = (off + bytes + 255) & ~(size_t)255;
        return p;
    };
    float*  deg    = (float*)carve((size_t)NNODES * 4);
    float*  dis    = (float*)carve((size_t)NNODES * 4);
    float*  nrm    = (float*)carve((size_t)NEDGES * 4);
    __bf16* xpad   = (__bf16*)carve((size_t)NNODES * K1PAD * 2);
    __bf16* w10p   = (__bf16*)carve((size_t)K1PAD * EMB0 * 2);
    __bf16* w11p   = (__bf16*)carve((size_t)K1PAD * EMB0 * 2);
    __bf16* w20p   = (__bf16*)carve((size_t)EMB0 * EMB1 * 2);
    __bf16* w21p   = (__bf16*)carve((size_t)EMB0 * EMB1 * 2);
    float*  acc    = (float*)carve((size_t)NNODES * K1PAD * 4);   // reused both layers
    __bf16* accb   = (__bf16*)carve((size_t)NNODES * K1PAD * 2);  // reused both layers
    __bf16* h1     = (__bf16*)carve((size_t)NNODES * EMB0 * 2);
    float*  h2     = (float*)carve((size_t)NNODES * EMB1 * 4);
    float*  ro     = (float*)carve((size_t)NNODES * 8 * 4);       // == feats [B, 928]
    float*  z      = (float*)carve((size_t)BATCH * NROI * 4);
    float*  mu     = (float*)carve((size_t)NROI * 4);
    float*  var    = (float*)carve((size_t)NROI * 4);
    (void)ws_size; (void)n_in; (void)in_sizes; (void)out_size;

    const int TB = 256;
    auto blk = [&](long long n) { return (unsigned)((n + TB - 1) / TB); };

    // ---- graph normalization ----------------------------------------------
    k_zero<<<blk(NNODES), TB, 0, stream>>>(deg, NNODES);
    k_deg<<<blk(NEDGES), TB, 0, stream>>>(esrc, eattr, deg, NEDGES);
    k_dis<<<blk(NNODES), TB, 0, stream>>>(deg, dis, NNODES);
    k_norm<<<blk(NEDGES), TB, 0, stream>>>(esrc, edst, eattr, dis, nrm, NEDGES);

    // ---- operand prep ------------------------------------------------------
    k_xpad<<<blk((long long)NNODES * K1PAD), TB, 0, stream>>>(x, xpad, NNODES);
    k_packw<<<blk((K1PAD / 32) * (EMB0 / 16) * 32), TB, 0, stream>>>(w1_0, w10p, NROI, K1PAD, EMB0);
    k_packw<<<blk((K1PAD / 32) * (EMB0 / 16) * 32), TB, 0, stream>>>(w1_1, w11p, NROI, K1PAD, EMB0);
    k_packw<<<blk((EMB0 / 32) * (EMB1 / 16) * 32), TB, 0, stream>>>(w2_0, w20p, EMB0, EMB0, EMB1);
    k_packw<<<blk((EMB0 / 32) * (EMB1 / 16) * 32), TB, 0, stream>>>(w2_1, w21p, EMB0, EMB0, EMB1);

    // ---- layer 1: tx1 scatter + fused dual WMMA GEMM + mish ---------------
    k_zero<<<blk((long long)NNODES * K1PAD), TB, 0, stream>>>(acc, NNODES * K1PAD);
    k_scatter_l1<<<blk((long long)NEDGES * 32), TB, 0, stream>>>(esrc, edst, nrm, x, acc);
    k_f2bf<<<blk((long long)NNODES * K1PAD), TB, 0, stream>>>(acc, accb, NNODES * K1PAD);
    {
        int waves = (NNODES / 16) * 2;                            // M tiles * groups
        k_gemm_dual<4, EMB0 / 16><<<(waves + 7) / 8, TB, 0, stream>>>(
            xpad, accb, w10p, w11p, b1, NNODES, h1, nullptr);
    }

    // ---- layer 2 -----------------------------------------------------------
    k_zero<<<blk((long long)NNODES * EMB0), TB, 0, stream>>>(acc, NNODES * EMB0);
    k_scatter_l2<<<blk((long long)NEDGES * 32), TB, 0, stream>>>(esrc, edst, nrm, h1, acc);
    k_f2bf<<<blk((long long)NNODES * EMB0), TB, 0, stream>>>(acc, accb, NNODES * EMB0);
    {
        int waves = NNODES / 16;                                  // M tiles * 1 group
        k_gemm_dual<4, EMB1 / 16><<<(waves + 7) / 8, TB, 0, stream>>>(
            h1, accb, w20p, w21p, b2, NNODES, nullptr, h2);
    }

    // ---- readout + FC1 + BN + FC2 -----------------------------------------
    k_readout<<<blk((long long)NNODES * 8), TB, 0, stream>>>(h2, w_ro, b_ro, ro);
    k_fc1<<<blk((long long)BATCH * NROI), TB, 0, stream>>>(ro, w_fc1, b_fc1, z);
    k_bnstats<<<NROI, 256, 0, stream>>>(z, mu, var);
    k_final<<<blk(BATCH), TB, 0, stream>>>(z, mu, var, bn_g, bn_b, w_fc2, b_fc2,
                                           (float*)d_out);
}
